// ComplexAttention_39513699123241
// MI455X (gfx1250) — matrix operand order repaired and checked by reading, not compiled, not run on hardware
//
#include <hip/hip_runtime.h>

// ---------------------------------------------------------------------------
// ComplexAttention for MI455X (gfx1250), wave32, WMMA bf16 16x16x32.
// B=4 H=8 T=2048 D=256.  All GEMMs run on v_wmma_f32_16x16x32_bf16.
// Workspace requirement: ~304 MB (9 bf16 [65536,256] buffers + bf16 weights).
// ---------------------------------------------------------------------------

typedef __attribute__((ext_vector_type(16))) __bf16 v16bf;
typedef __attribute__((ext_vector_type(8)))  float  v8f;

#define D_MODEL 256
#define SEQ_T   2048
#define NBH     32
#define NROWS   (NBH * SEQ_T)                 // 65536 token rows
static const size_t MATE = (size_t)NROWS * D_MODEL;  // 16,777,216 elems / buffer

union BF16x16 {            // one 16x32 A-fragment or 32x16 B-fragment per lane
    v16bf v;
    uint4 q[2];
};

__device__ inline v8f wmma_bf16(v16bf a, v16bf b, v8f c) {
    // (neg_a, A, neg_b, B, c_mod, C, reuse_a, reuse_b)
    return __builtin_amdgcn_wmma_f32_16x16x32_bf16(false, a, false, b, (short)0, c,
                                                   false, false);
}

// A fragment (M=16,K=32) from row-major bf16 [*, ld]:
//   lanes 0-15: M=lane,   K chunks {kb+0..7, kb+16..23}
//   lanes 16-31: M=lane-16, K chunks {kb+8..15, kb+24..31}
__device__ inline BF16x16 load_a_bf16(const __bf16* base, int ld, int kb,
                                      int hf, int mrow) {
    BF16x16 a;
    const __bf16* p = base + (size_t)mrow * ld + kb + hf * 8;
    a.q[0] = *(const uint4*)(p);
    a.q[1] = *(const uint4*)(p + 16);
    return a;
}

// Same A fragment, sourced from f32 with on-the-fly bf16 conversion.
__device__ inline BF16x16 load_a_f32(const float* base, int ld, int kb,
                                     int hf, int mrow) {
    const float* p = base + (size_t)mrow * ld + kb + hf * 8;
    float4 f0 = *(const float4*)(p);
    float4 f1 = *(const float4*)(p + 4);
    float4 f2 = *(const float4*)(p + 16);
    float4 f3 = *(const float4*)(p + 20);
    BF16x16 a;
    a.v[0]=(__bf16)f0.x;  a.v[1]=(__bf16)f0.y;  a.v[2]=(__bf16)f0.z;  a.v[3]=(__bf16)f0.w;
    a.v[4]=(__bf16)f1.x;  a.v[5]=(__bf16)f1.y;  a.v[6]=(__bf16)f1.z;  a.v[7]=(__bf16)f1.w;
    a.v[8]=(__bf16)f2.x;  a.v[9]=(__bf16)f2.y;  a.v[10]=(__bf16)f2.z; a.v[11]=(__bf16)f2.w;
    a.v[12]=(__bf16)f3.x; a.v[13]=(__bf16)f3.y; a.v[14]=(__bf16)f3.z; a.v[15]=(__bf16)f3.w;
    return a;
}

// B fragment (K=32,N=16) where B[k,n] = src[nbase+n][kb+k], src row-major [*, ld].
// lane L: n = L%16, K chunks {kb+(L/16)*8+0..7, kb+16+(L/16)*8+0..7} (matches A).
__device__ inline BF16x16 load_b(const __bf16* src, int ld, int kb, int nbase,
                                 int hf, int mrow) {
    BF16x16 b;
    const __bf16* p = src + (size_t)(nbase + mrow) * ld + kb + hf * 8;
    b.q[0] = *(const uint4*)(p);
    b.q[1] = *(const uint4*)(p + 16);
    return b;
}

// --------------------------- weight f32 -> bf16 ----------------------------
__global__ void cvt_kernel(const float* __restrict__ src, __bf16* __restrict__ dst,
                           int n) {
    int i = blockIdx.x * 256 + threadIdx.x;
    if (i < n) dst[i] = (__bf16)src[i];
}

// --------------------- complex linear: y = clinear(x, W) -------------------
// grid = (NROWS/16, D/16), block = 32 (one wave). transpose!=0 stores y as [bh][D][T].
__global__ void proj_kernel(const float* __restrict__ xr, const float* __restrict__ xi,
                            const __bf16* __restrict__ wr, const __bf16* __restrict__ wi,
                            const float* __restrict__ br, const float* __restrict__ bi,
                            __bf16* __restrict__ yr, __bf16* __restrict__ yi,
                            int transpose) {
    int rowTile = blockIdx.x;
    int nt      = blockIdx.y;
    int ln = threadIdx.x & 31, hf = ln >> 4, mrow = ln & 15;

    v8f Arr = {0,0,0,0,0,0,0,0}, Aii = Arr, Air = Arr, Ari = Arr;
    const size_t xbase = (size_t)rowTile * 16 * D_MODEL;

#pragma unroll
    for (int ds = 0; ds < 8; ++ds) {
        int kb = ds * 32;
        BF16x16 axr = load_a_f32(xr + xbase, D_MODEL, kb, hf, mrow);
        BF16x16 axi = load_a_f32(xi + xbase, D_MODEL, kb, hf, mrow);
        BF16x16 bwr = load_b(wr, D_MODEL, kb, nt * 16, hf, mrow);
        BF16x16 bwi = load_b(wi, D_MODEL, kb, nt * 16, hf, mrow);
        Arr = wmma_bf16(axr.v, bwr.v, Arr);
        Aii = wmma_bf16(axi.v, bwi.v, Aii);
        Air = wmma_bf16(axi.v, bwr.v, Air);
        Ari = wmma_bf16(axr.v, bwi.v, Ari);
    }

#pragma unroll
    for (int r = 0; r < 8; ++r) {
        int row = rowTile * 16 + r + hf * 8;
        int col = nt * 16 + mrow;
        float vr_ = Arr[r] - Aii[r] + br[col];
        float vi_ = Air[r] + Ari[r] + bi[col];
        if (transpose) {  // [bh][D][T] for V so ctx GEMM B-fragments are contiguous
            int bh = row >> 11, t = row & 2047;
            size_t o = ((size_t)bh * D_MODEL + col) * SEQ_T + t;
            yr[o] = (__bf16)vr_;
            yi[o] = (__bf16)vi_;
        } else {          // [row][D]
            size_t o = (size_t)row * D_MODEL + col;
            yr[o] = (__bf16)vr_;
            yi[o] = (__bf16)vi_;
        }
    }
}

// ------------------------- flash attention core ----------------------------
// grid = (32 bh, 128 q-tiles), block = 32 (one wave, 16 q-rows, full D ctx).
__global__ void attn_kernel(const __bf16* __restrict__ Qr, const __bf16* __restrict__ Qi,
                            const __bf16* __restrict__ Kr, const __bf16* __restrict__ Ki,
                            const __bf16* __restrict__ Vtr, const __bf16* __restrict__ Vti,
                            __bf16* __restrict__ Cr, __bf16* __restrict__ Ci) {
    int bh = blockIdx.x;
    int qt = blockIdx.y;
    int ln = threadIdx.x & 31, hf = ln >> 4, mrow = ln & 15;

    __shared__ __attribute__((aligned(16))) __bf16 sQ[2][16 * D_MODEL];  // 16 KB
    __shared__ __attribute__((aligned(16))) __bf16 sP[16 * 32];          // 1 KB

    const size_t qoff = (size_t)(bh * SEQ_T + qt * 16) * D_MODEL;
    {   // 16 rows x 256 cols of Q_r / Q_i are contiguous: plain b128 copies
        const uint4* gr = (const uint4*)(Qr + qoff);
        const uint4* gi = (const uint4*)(Qi + qoff);
        uint4* sr = (uint4*)sQ[0];
        uint4* si = (uint4*)sQ[1];
#pragma unroll
        for (int i = 0; i < 16; ++i) {
            sr[i * 32 + ln] = gr[i * 32 + ln];
            si[i * 32 + ln] = gi[i * 32 + ln];
        }
    }
    __syncthreads();

    float m[8], l[8];
    v8f Or[16], Oi[16];
#pragma unroll
    for (int r = 0; r < 8; ++r) { m[r] = -3.0e38f; l[r] = 0.0f; }
    v8f zero = {0,0,0,0,0,0,0,0};
#pragma unroll
    for (int n = 0; n < 16; ++n) { Or[n] = zero; Oi[n] = zero; }

    const __bf16* vrb = Vtr + (size_t)bh * D_MODEL * SEQ_T;
    const __bf16* vib = Vti + (size_t)bh * D_MODEL * SEQ_T;

    for (int kt = 0; kt < SEQ_T / 32; ++kt) {
        const __bf16* krow = Kr + (size_t)(bh * SEQ_T + kt * 32) * D_MODEL;
        const __bf16* kirow = Ki + (size_t)(bh * SEQ_T + kt * 32) * D_MODEL;

        v8f S0 = zero, S1 = zero;   // scores for keys 0..15 / 16..31 of tile
#pragma unroll
        for (int ds = 0; ds < 8; ++ds) {
            int kb = ds * 32;
            BF16x16 aqr = load_a_bf16(sQ[0], D_MODEL, kb, hf, mrow);
            BF16x16 aqi = load_a_bf16(sQ[1], D_MODEL, kb, hf, mrow);
            BF16x16 b0r = load_b(krow,  D_MODEL, kb, 0,  hf, mrow);
            BF16x16 b1r = load_b(krow,  D_MODEL, kb, 16, hf, mrow);
            BF16x16 b0i = load_b(kirow, D_MODEL, kb, 0,  hf, mrow);
            BF16x16 b1i = load_b(kirow, D_MODEL, kb, 16, hf, mrow);
            S0 = wmma_bf16(aqr.v, b0r.v, S0);
            S0 = wmma_bf16(aqi.v, b0i.v, S0);
            S1 = wmma_bf16(aqr.v, b1r.v, S1);
            S1 = wmma_bf16(aqi.v, b1i.v, S1);
        }

        // online softmax: each 16-lane group holds one row per accumulator slot
        float alpha[8];
#pragma unroll
        for (int r = 0; r < 8; ++r) {
            float s0 = S0[r] * 0.0625f;        // 1/sqrt(256)
            float s1 = S1[r] * 0.0625f;
            S0[r] = s0; S1[r] = s1;
            float v = fmaxf(s0, s1);
            v = fmaxf(v, __shfl_xor(v, 1, 16));
            v = fmaxf(v, __shfl_xor(v, 2, 16));
            v = fmaxf(v, __shfl_xor(v, 4, 16));
            v = fmaxf(v, __shfl_xor(v, 8, 16));
            float mn = fmaxf(m[r], v);
            alpha[r] = __expf(m[r] - mn);
            m[r] = mn;
            float p0 = __expf(s0 - mn);
            float p1 = __expf(s1 - mn);
            S0[r] = p0; S1[r] = p1;
            float rs = p0 + p1;
            rs += __shfl_xor(rs, 1, 16);
            rs += __shfl_xor(rs, 2, 16);
            rs += __shfl_xor(rs, 4, 16);
            rs += __shfl_xor(rs, 8, 16);
            l[r] = l[r] * alpha[r] + rs;
        }
#pragma unroll
        for (int n = 0; n < 16; ++n)
#pragma unroll
            for (int r = 0; r < 8; ++r) {
                Or[n][r] *= alpha[r];
                Oi[n][r] *= alpha[r];
            }

        // D-layout P -> row-major LDS -> A-layout P (16x32 bf16)
        __syncthreads();
#pragma unroll
        for (int r = 0; r < 8; ++r) {
            int row = r + hf * 8;
            sP[row * 32 + mrow]      = (__bf16)S0[r];
            sP[row * 32 + 16 + mrow] = (__bf16)S1[r];
        }
        __syncthreads();
        BF16x16 ap = load_a_bf16(sP, 32, 0, hf, mrow);

        // ctx += P @ V  (V stored transposed [D][T] => contiguous B loads)
#pragma unroll
        for (int n = 0; n < 16; ++n) {
            BF16x16 bvr = load_b(vrb, SEQ_T, kt * 32, n * 16, hf, mrow);
            BF16x16 bvi = load_b(vib, SEQ_T, kt * 32, n * 16, hf, mrow);
            Or[n] = wmma_bf16(ap.v, bvr.v, Or[n]);
            Oi[n] = wmma_bf16(ap.v, bvi.v, Oi[n]);
        }
    }

    float inv[8];
#pragma unroll
    for (int r = 0; r < 8; ++r) inv[r] = 1.0f / l[r];
    const size_t obase = qoff;
#pragma unroll
    for (int n = 0; n < 16; ++n)
#pragma unroll
        for (int r = 0; r < 8; ++r) {
            int row = r + hf * 8;
            int col = n * 16 + mrow;
            Cr[obase + (size_t)row * D_MODEL + col] = (__bf16)(Or[n][r] * inv[r]);
            Ci[obase + (size_t)row * D_MODEL + col] = (__bf16)(Oi[n][r] * inv[r]);
        }
}

// -------------- output projection + magnitude (pre-gate) -------------------
__global__ void oproj_kernel(const __bf16* __restrict__ cr, const __bf16* __restrict__ ci,
                             const __bf16* __restrict__ wr, const __bf16* __restrict__ wi,
                             const float* __restrict__ br, const float* __restrict__ bi,
                             float* __restrict__ out_r, float* __restrict__ out_i,
                             __bf16* __restrict__ mag) {
    int rowTile = blockIdx.x;
    int nt      = blockIdx.y;
    int ln = threadIdx.x & 31, hf = ln >> 4, mrow = ln & 15;

    v8f Arr = {0,0,0,0,0,0,0,0}, Aii = Arr, Air = Arr, Ari = Arr;
    const size_t xbase = (size_t)rowTile * 16 * D_MODEL;
#pragma unroll
    for (int ds = 0; ds < 8; ++ds) {
        int kb = ds * 32;
        BF16x16 acr = load_a_bf16(cr + xbase, D_MODEL, kb, hf, mrow);
        BF16x16 aci = load_a_bf16(ci + xbase, D_MODEL, kb, hf, mrow);
        BF16x16 bwr = load_b(wr, D_MODEL, kb, nt * 16, hf, mrow);
        BF16x16 bwi = load_b(wi, D_MODEL, kb, nt * 16, hf, mrow);
        Arr = wmma_bf16(acr.v, bwr.v, Arr);
        Aii = wmma_bf16(aci.v, bwi.v, Aii);
        Air = wmma_bf16(aci.v, bwr.v, Air);
        Ari = wmma_bf16(acr.v, bwi.v, Ari);
    }
#pragma unroll
    for (int r = 0; r < 8; ++r) {
        int row = rowTile * 16 + r + hf * 8;
        int col = nt * 16 + mrow;
        size_t o = (size_t)row * D_MODEL + col;
        float vr_ = Arr[r] - Aii[r] + br[col];
        float vi_ = Air[r] + Ari[r] + bi[col];
        out_r[o] = vr_;
        out_i[o] = vi_;
        mag[o] = (__bf16)sqrtf(vr_ * vr_ + vi_ * vi_ + 1e-8f);
    }
}

// ------------------ gate GEMM + sigmoid + apply in place -------------------
__global__ void gate_kernel(const __bf16* __restrict__ mag,
                            const __bf16* __restrict__ gw, const float* __restrict__ gb,
                            float* __restrict__ out_r, float* __restrict__ out_i) {
    int rowTile = blockIdx.x;
    int nt      = blockIdx.y;
    int ln = threadIdx.x & 31, hf = ln >> 4, mrow = ln & 15;

    v8f acc = {0,0,0,0,0,0,0,0};
    const size_t xbase = (size_t)rowTile * 16 * D_MODEL;
#pragma unroll
    for (int ds = 0; ds < 8; ++ds) {
        int kb = ds * 32;
        BF16x16 am = load_a_bf16(mag + xbase, D_MODEL, kb, hf, mrow);
        BF16x16 bw = load_b(gw, D_MODEL, kb, nt * 16, hf, mrow);
        acc = wmma_bf16(am.v, bw.v, acc);
    }
#pragma unroll
    for (int r = 0; r < 8; ++r) {
        int row = rowTile * 16 + r + hf * 8;
        int col = nt * 16 + mrow;
        size_t o = (size_t)row * D_MODEL + col;
        float g = 1.0f / (1.0f + __expf(-(acc[r] + gb[col])));
        out_r[o] *= g;   // each (row,col) owned by exactly this tile -> no race
        out_i[o] *= g;
    }
}

// ---------------------------------------------------------------------------
extern "C" void kernel_launch(void* const* d_in, const int* in_sizes, int n_in,
                              void* d_out, int out_size, void* d_ws, size_t ws_size,
                              hipStream_t stream) {
    const float* q_in_r  = (const float*)d_in[0];
    const float* q_in_i  = (const float*)d_in[1];
    const float* kv_in_r = (const float*)d_in[2];
    const float* kv_in_i = (const float*)d_in[3];
    const float* q_wr = (const float*)d_in[4];
    const float* q_wi = (const float*)d_in[5];
    const float* q_br = (const float*)d_in[6];
    const float* q_bi = (const float*)d_in[7];
    const float* k_wr = (const float*)d_in[8];
    const float* k_wi = (const float*)d_in[9];
    const float* k_br = (const float*)d_in[10];
    const float* k_bi = (const float*)d_in[11];
    const float* v_wr = (const float*)d_in[12];
    const float* v_wi = (const float*)d_in[13];
    const float* v_br = (const float*)d_in[14];
    const float* v_bi = (const float*)d_in[15];
    const float* o_wr = (const float*)d_in[16];
    const float* o_wi = (const float*)d_in[17];
    const float* o_br = (const float*)d_in[18];
    const float* o_bi = (const float*)d_in[19];
    const float* gate_w = (const float*)d_in[20];
    const float* gate_b = (const float*)d_in[21];

    // workspace carve-up (bf16 elements)
    __bf16* ws  = (__bf16*)d_ws;
    __bf16* Qr  = ws;
    __bf16* Qi  = Qr  + MATE;
    __bf16* Kr  = Qi  + MATE;
    __bf16* Ki  = Kr  + MATE;
    __bf16* Vtr = Ki  + MATE;   // [bh][D][T]
    __bf16* Vti = Vtr + MATE;
    __bf16* Cr  = Vti + MATE;
    __bf16* Ci  = Cr  + MATE;
    __bf16* Mag = Ci  + MATE;
    __bf16* W   = Mag + MATE;   // 9 x 256x256 bf16 weight copies
    const int WN = D_MODEL * D_MODEL;
    __bf16 *bq_r = W,          *bq_i = W + WN,     *bk_r = W + 2*WN, *bk_i = W + 3*WN;
    __bf16 *bv_r = W + 4*WN,   *bv_i = W + 5*WN,   *bo_r = W + 6*WN, *bo_i = W + 7*WN;
    __bf16 *bg_w = W + 8*WN;

    // 1) weights -> bf16
    dim3 cg((WN + 255) / 256), cb(256);
    cvt_kernel<<<cg, cb, 0, stream>>>(q_wr, bq_r, WN);
    cvt_kernel<<<cg, cb, 0, stream>>>(q_wi, bq_i, WN);
    cvt_kernel<<<cg, cb, 0, stream>>>(k_wr, bk_r, WN);
    cvt_kernel<<<cg, cb, 0, stream>>>(k_wi, bk_i, WN);
    cvt_kernel<<<cg, cb, 0, stream>>>(v_wr, bv_r, WN);
    cvt_kernel<<<cg, cb, 0, stream>>>(v_wi, bv_i, WN);
    cvt_kernel<<<cg, cb, 0, stream>>>(o_wr, bo_r, WN);
    cvt_kernel<<<cg, cb, 0, stream>>>(o_wi, bo_i, WN);
    cvt_kernel<<<cg, cb, 0, stream>>>(gate_w, bg_w, WN);

    // 2) Q/K/V complex projections (V transposed for ctx GEMM)
    dim3 pg(NROWS / 16, D_MODEL / 16), pb(32);
    proj_kernel<<<pg, pb, 0, stream>>>(q_in_r,  q_in_i,  bq_r, bq_i, q_br, q_bi, Qr,  Qi,  0);
    proj_kernel<<<pg, pb, 0, stream>>>(kv_in_r, kv_in_i, bk_r, bk_i, k_br, k_bi, Kr,  Ki,  0);
    proj_kernel<<<pg, pb, 0, stream>>>(kv_in_r, kv_in_i, bv_r, bv_i, v_br, v_bi, Vtr, Vti, 1);

    // 3) flash attention
    attn_kernel<<<dim3(NBH, SEQ_T / 16), 32, 0, stream>>>(Qr, Qi, Kr, Ki, Vtr, Vti, Cr, Ci);

    // 4) output projection (+ magnitude), pre-gate results straight into d_out
    float* out_r = (float*)d_out;
    float* out_i = out_r + MATE;
    oproj_kernel<<<pg, pb, 0, stream>>>(Cr, Ci, bo_r, bo_i, o_br, o_bi, out_r, out_i, Mag);

    // 5) gate GEMM + sigmoid, applied in place
    gate_kernel<<<pg, pb, 0, stream>>>(Mag, bg_w, gate_b, out_r, out_i);
}
